// GaussianRenderer_78993038508224
// MI455X (gfx1250) — matrix operand order repaired and checked by reading, not compiled, not run on hardware
//
#include <hip/hip_runtime.h>
#include <hip/hip_bf16.h>

// ---------------------------------------------------------------------------
// Gaussian splat renderer for MI455X (gfx1250, wave32).
//
// Phase 1 (gs_prep, 2048 threads): per-gaussian factorization.
//   exponent(px,py) = -0.5*dx^T Sigma^-1 dx  ==  f(pixel) . g(gaussian)
//   with f = [px^2, py^2, px*py, px, py, 1] and g the 6 derived coefficients.
//   Also: opacity = sigmoid(p8), colors = sigmoid(p5..7) stored as f16 in a
//   channel-major [B][16][N] matrix (ch 0..2 = RGB, ch 3 = 1.0 for the
//   weights-sum column, ch 4..15 = 0 padding for the 16-wide WMMA N dim).
//
// Phase 2 (gs_render): each wave owns 16 pixels. Loop over gaussians in
//   chunks of 32 (WMMA K): f32 VALU evaluates the 6-term dot + clip + v_exp
//   per (pixel,gaussian), packs f16 weights directly into a contiguous-K A-fragment
//   layout, and v_wmma_f32_16x16x32_f16 accumulates [16pix x 32g] x [32g x
//   16ch] into an f32 [16 x 16] tile. Column 3 of the accumulator is the
//   weights sum -> final divide is free of any extra reduction.
// ---------------------------------------------------------------------------

typedef __attribute__((ext_vector_type(16))) _Float16 v16h;
typedef __attribute__((ext_vector_type(8)))  float    v8f;

#define B_BATCH 2
#define H_IN    32
#define W_IN    32
#define NG      (H_IN * W_IN)      // 1024 gaussians per batch
#define H_OUT   128
#define W_OUT   128
#define HW      (H_OUT * W_OUT)    // 16384 pixels
#define WPB     4                  // waves per block (128 threads, wave32)

__global__ __launch_bounds__(256)
void gs_prep(const float* __restrict__ params,
             float* __restrict__ gcoef,        // [B][NG][8] f32
             _Float16* __restrict__ colT) {    // [B][16][NG] f16 channel-major
  const int idx = blockIdx.x * blockDim.x + threadIdx.x;
  if (idx >= B_BATCH * NG) return;
  const int b = idx / NG;
  const int n = idx - b * NG;
  const int i = n / W_IN;          // row in h_in
  const int j = n - i * W_IN;      // col in w_in

  const float* pp = params + (size_t)b * 9 * NG + n;   // [B][9][NG]
  const float p0 = pp[0 * NG], p1 = pp[1 * NG], p2 = pp[2 * NG];
  const float p3 = pp[3 * NG], p4 = pp[4 * NG], p5 = pp[5 * NG];
  const float p6 = pp[6 * NG], p7 = pp[7 * NG], p8 = pp[8 * NG];

  // base_grid: linspace(-1+1/w, 1-1/w, w) has step 2/w -> -1 + (2j+1)/w
  const float bx = -1.0f + (2.0f * j + 1.0f) / W_IN;
  const float by = -1.0f + (2.0f * i + 1.0f) / H_IN;
  const float mux = bx + tanhf(p0) * (2.0f / H_IN);
  const float muy = by + tanhf(p1) * (2.0f / H_IN);

  // cov = L L^T + 1e-5 I, L = [[e^p2, 0],[p3, e^p4]]
  const float v1x = expf(p2), v2x = p3, v2y = expf(p4);
  const float c00 = v1x * v1x + 1e-5f;
  const float c01 = v1x * v2x;
  const float c11 = v2x * v2x + v2y * v2y + 1e-5f;
  const float id  = 1.0f / (c00 * c11 - c01 * c01);
  const float A  =  c11 * id;       // Sigma_inv[0][0]
  const float Bq = -c01 * id;       // Sigma_inv[0][1]
  const float C  =  c00 * id;       // Sigma_inv[1][1]

  // exponent = f.g with f = [x^2, y^2, xy, x, y, 1]
  float g[8];
  g[0] = -0.5f * A;
  g[1] = -0.5f * C;
  g[2] = -Bq;
  g[3] = A * mux + Bq * muy;
  g[4] = Bq * mux + C * muy;
  g[5] = -0.5f * (A * mux * mux + 2.0f * Bq * mux * muy + C * muy * muy);
  g[6] = 1.0f / (1.0f + expf(-p8));   // opacity
  g[7] = 0.0f;

  float4* go = (float4*)(gcoef + (size_t)idx * 8);
  go[0] = make_float4(g[0], g[1], g[2], g[3]);
  go[1] = make_float4(g[4], g[5], g[6], g[7]);

  const float r  = 1.0f / (1.0f + expf(-p5));
  const float gg = 1.0f / (1.0f + expf(-p6));
  const float bb = 1.0f / (1.0f + expf(-p7));
  _Float16* cbase = colT + (size_t)b * 16 * NG + n;
  cbase[0 * NG] = (_Float16)r;
  cbase[1 * NG] = (_Float16)gg;
  cbase[2 * NG] = (_Float16)bb;
  cbase[3 * NG] = (_Float16)1.0f;     // weights-sum column
#pragma unroll
  for (int c = 4; c < 16; ++c) cbase[c * NG] = (_Float16)0.0f;
}

__global__ __launch_bounds__(32 * WPB)
void gs_render(const float* __restrict__ gcoef,
               const _Float16* __restrict__ colT,
               float* __restrict__ out) {
  const int lane = threadIdx.x & 31;
  const int wave = threadIdx.x >> 5;
  const int tile = blockIdx.x * WPB + wave;        // one 16-pixel tile / wave
  const int tilesPerBatch = HW / 16;
  const int b  = tile / tilesPerBatch;
  const int p0 = (tile - b * tilesPerBatch) * 16;
  const int half = lane >> 4;                      // lane group (ISA layouts)
  const int col  = lane & 15;                      // B/D column, A row

  // pixel_coordinates: linspace(-1, 1, n) -> -1 + 2*i/(n-1)
  const int   pix = p0 + col;                      // A-matrix row M = lane&15
  const int   yi  = pix / W_OUT;
  const int   xi  = pix - yi * W_OUT;
  const float px  = -1.0f + 2.0f * xi / (W_OUT - 1);
  const float py  = -1.0f + 2.0f * yi / (H_OUT - 1);
  const float f0 = px * px, f1 = py * py, f2 = px * py;

  const float*    gb = gcoef + (size_t)b * NG * 8;
  const _Float16* cb = colT + (size_t)b * 16 * NG + (size_t)col * NG;

  v8f acc = {};
#pragma unroll 1
  for (int chunk = 0; chunk < NG; chunk += 32) {
    // Build A fragment: 16x32 f16, row M = lane&15, contiguous K order
    //   matching B: element jj holds K = half*16 + jj
    v16h a;
#pragma unroll
    for (int jj = 0; jj < 16; ++jj) {
      const int K = half * 16 + jj;
      const float4* gp = (const float4*)(gb + (size_t)(chunk + K) * 8);
      const float4 g0 = gp[0];
      const float4 g1 = gp[1];
      float e = f0 * g0.x + f1 * g0.y + f2 * g0.z + px * g0.w + py * g1.x + g1.y;
      e = fminf(fmaxf(e, -20.0f), 20.0f);          // clip BEFORE opacity (ref)
      a[jj] = (_Float16)(g1.z * __expf(e));        // w = opacity * exp(e)
    }
    // B fragment: 32x16 f16, column N = lane&15; lanes 0-15 hold K=0..15,
    // lanes 16-31 hold K=16..31 -> one aligned 32B contiguous f16 load.
    const v16h bfrag = *(const v16h*)(cb + chunk + half * 16);
    acc = __builtin_amdgcn_wmma_f32_16x16x32_f16(
        /*neg_a=*/false, a, /*neg_b=*/false, bfrag,
        /*c_mod=*/(short)0, acc, /*reuse_a=*/false, /*reuse_b=*/false);
  }

  // D layout: lane holds column N=lane&15; VGPR r holds row M = r + 8*half.
  __shared__ float red[WPB][16][4];
  if (col < 4) {
#pragma unroll
    for (int r = 0; r < 8; ++r) red[wave][r + half * 8][col] = acc[r];
  }
  __syncthreads();
  if (lane < 16) {
    const float inv = 1.0f / (red[wave][lane][3] + 1e-8f);  // 1/(wsum+eps)
    const int   p   = p0 + lane;
    float* ob = out + (size_t)b * 3 * HW + p;
    ob[0]      = red[wave][lane][0] * inv;
    ob[HW]     = red[wave][lane][1] * inv;
    ob[2 * HW] = red[wave][lane][2] * inv;
  }
}

extern "C" void kernel_launch(void* const* d_in, const int* in_sizes, int n_in,
                              void* d_out, int out_size, void* d_ws, size_t ws_size,
                              hipStream_t stream) {
  (void)in_sizes; (void)n_in; (void)out_size; (void)ws_size;
  const float* params = (const float*)d_in[0];   // [2,9,32,32] f32
  // d_in[1]/d_in[2] are h_out/w_out scalars (128,128) — shapes fixed by ref.
  float*    gcoef = (float*)d_ws;                                  // 64 KB
  _Float16* colT  = (_Float16*)((char*)d_ws +
                                (size_t)B_BATCH * NG * 8 * sizeof(float)); // 64 KB
  float* out = (float*)d_out;                    // [2,3,128,128] f32

  gs_prep<<<(B_BATCH * NG + 255) / 256, 256, 0, stream>>>(params, gcoef, colT);

  const int tiles = B_BATCH * (HW / 16);         // 2048 wave-tiles
  gs_render<<<tiles / WPB, 32 * WPB, 0, stream>>>(gcoef, colT, out);
}
